// SelfAttention_50843822850169
// MI455X (gfx1250) — matrix operand order repaired and checked by reading, not compiled, not run on hardware
//
#include <hip/hip_runtime.h>
#include <stdint.h>
#include <stddef.h>

// Problem constants (match reference): B=2, S=2048, D=1024, H=16, HD=64
#define B_N   2
#define S_N   2048
#define D_N   1024
#define H_N   16
#define HD_N  64
#define M_TOT (B_N * S_N)          // 4096 rows for the projection GEMMs
#define ATT_SCALE 0.125f           // 1/sqrt(64)
#define NEG_BIG (-3.0e38f)

typedef __bf16 v8bf  __attribute__((ext_vector_type(8)));
typedef __bf16 v16bf __attribute__((ext_vector_type(16)));
typedef float  v8f   __attribute__((ext_vector_type(8)));

static __device__ __forceinline__ uint16_t f32_bf16(float f) {
  union { float f; uint32_t u; } c; c.f = f;
  uint32_t u = c.u;
  u += 0x7FFFu + ((u >> 16) & 1u);   // round-to-nearest-even
  return (uint16_t)(u >> 16);
}

// Build a 16-wide bf16 fragment from two 8-element (16-byte) loads.
static __device__ __forceinline__ v16bf frag_cat(const uint16_t* p0, const uint16_t* p1) {
  v8bf lo = *(const v8bf*)p0;
  v8bf hi = *(const v8bf*)p1;
  return __builtin_shufflevector(lo, hi, 0,1,2,3,4,5,6,7,8,9,10,11,12,13,14,15);
}

// Async global->LDS 16B copy (CDNA5, tracked by ASYNCcnt).
static __device__ __forceinline__ void async_cp16(uint32_t lds_off, uint32_t gbl_off,
                                                  const uint16_t* base) {
  asm volatile("global_load_async_to_lds_b128 %0, %1, %2"
               :: "v"(lds_off), "v"(gbl_off), "s"(base) : "memory");
}
static __device__ __forceinline__ void wait_async0() {
  asm volatile("s_wait_asynccnt 0x0" ::: "memory");
}

// ---------------------------------------------------------------------------
// Kernel 0: f32 -> bf16 bits, optional scale (folds softmax scale into Wq)
// ---------------------------------------------------------------------------
__global__ void cvt_bf16_k(const float* __restrict__ src, uint16_t* __restrict__ dst,
                           int n, float scale) {
  int i = blockIdx.x * blockDim.x + threadIdx.x;
  if (i < n) dst[i] = f32_bf16(src[i] * scale);
}

// ---------------------------------------------------------------------------
// Kernel 1: Y[m,n] = sum_k X[m,k] * W[n,k]  (x @ W^T), bf16 in, f32 acc.
// Block = 256 thr (8 waves) -> 128x128 output tile; wave = 32x64 (2 M x 4 N
// WMMA tiles). Per 32-deep k-step: X(128x32) and W(128x32) bf16 tiles staged
// in LDS via GLOBAL_LOAD_ASYNC_TO_LDS_B128, double-buffered so the copy for
// step k+1 overlaps the WMMAs of step k.
// mode 0: bf16 per-head [b][h][s][d] (Q,K); 1: transposed [b][h][d][s] (V);
// mode 2: f32 row-major (final projection)
// ---------------------------------------------------------------------------
__global__ __launch_bounds__(256) void gemm_xwT_k(
    const uint16_t* __restrict__ X, const uint16_t* __restrict__ W,
    uint16_t* __restrict__ Ybf, float* __restrict__ Yf, int mode)
{
  __shared__ uint16_t lds[2][2][128 * 32];   // [buf][X/W][row*32 + k], 32 KB

  const int tid  = threadIdx.x;
  const int lane = tid & 31;
  const int wave = tid >> 5;
  const int ln   = lane & 15;
  const int hf   = lane >> 4;
  const int wm   = wave & 3;    // M strip (32 rows each)
  const int wn   = wave >> 2;   // N strip (64 cols each)
  const int mBase = blockIdx.x * 128;
  const int nBase = blockIdx.y * 128;

  // Async tile stage: 512 x 16B chunks per tile, 2 chunks/thread/tile.
  auto stage = [&](int buf, int k) {
    const uint32_t ldsX = (uint32_t)(uintptr_t)&lds[buf][0][0];
    const uint32_t ldsW = (uint32_t)(uintptr_t)&lds[buf][1][0];
#pragma unroll
    for (int i = 0; i < 2; ++i) {
      const int e = tid + i * 256;      // 0..511
      const int row = e >> 2;           // 0..127
      const int qd  = e & 3;            // 16B quad within 64B row
      async_cp16(ldsX + (uint32_t)(row * 64 + qd * 16),
                 (uint32_t)(((mBase + row) * D_N + k) * 2 + qd * 16), X);
      async_cp16(ldsW + (uint32_t)(row * 64 + qd * 16),
                 (uint32_t)(((nBase + row) * D_N + k) * 2 + qd * 16), W);
    }
  };

  v8f acc[2][4] = {};

  stage(0, 0);
  wait_async0();
  __syncthreads();

  for (int kt = 0; kt < D_N / 32; ++kt) {
    const int buf = kt & 1;
    if (kt + 1 < D_N / 32) stage(buf ^ 1, (kt + 1) * 32);   // overlap DMA w/ math

    const uint16_t* xs  = &lds[buf][0][0];
    const uint16_t* wsd = &lds[buf][1][0];

    v16bf bfrag[4];
#pragma unroll
    for (int j = 0; j < 4; ++j) {
      const uint16_t* wrow = wsd + (wn * 64 + j * 16 + ln) * 32 + hf * 16;
      bfrag[j] = frag_cat(wrow, wrow + 8);
    }
#pragma unroll
    for (int mt = 0; mt < 2; ++mt) {
      const uint16_t* xrow = xs + (wm * 32 + mt * 16 + ln) * 32;
      v16bf a = frag_cat(xrow + hf * 16, xrow + hf * 16 + 8);
#pragma unroll
      for (int j = 0; j < 4; ++j)
        acc[mt][j] = __builtin_amdgcn_wmma_f32_16x16x32_bf16(
            false, a, false, bfrag[j], (short)0, acc[mt][j], false, false);
    }

    wait_async0();
    __syncthreads();
  }

#pragma unroll
  for (int mt = 0; mt < 2; ++mt) {
#pragma unroll
    for (int j = 0; j < 4; ++j) {
#pragma unroll
      for (int r = 0; r < 8; ++r) {
        const int m = mBase + wm * 32 + mt * 16 + r + hf * 8;
        const int n = nBase + wn * 64 + j * 16 + ln;
        const float y = acc[mt][j][r];
        if (mode == 2) {
          Yf[(size_t)m * D_N + n] = y;
        } else {
          const int b = m / S_N, s = m % S_N, h = n / HD_N, d = n % HD_N;
          if (mode == 0)
            Ybf[(((size_t)(b * H_N + h)) * S_N + s) * HD_N + d] = f32_bf16(y);
          else
            Ybf[(((size_t)(b * H_N + h)) * HD_N + d) * S_N + s] = f32_bf16(y);
        }
      }
    }
  }
}

// ---------------------------------------------------------------------------
// Kernel 2: causal flash attention, block-cooperative.
// Block = 4 waves = 4 consecutive 16-row q-tiles (64 q rows) of one (b,h).
// The 32-key K (32x64) and V^T (64x32) bf16 tiles are shared by all 4 waves:
// staged in LDS with GLOBAL_LOAD_ASYNC_TO_LDS_B128, double-buffered so the
// copy of tile i+1 overlaps the math of tile i. Waves whose causal extent
// ended keep staging/barriers (uniform branch, EXEC all-1s for WMMA).
// Per tile per wave: 4x WMMA for S = Q K^T, online softmax (f32, half-wave
// shfl reductions), P transposed C->A layout via per-wave 1KB LDS bounce,
// 4x WMMA for O += P V.
// ---------------------------------------------------------------------------
__global__ __launch_bounds__(128) void flash_attn_k(
    const uint16_t* __restrict__ Q, const uint16_t* __restrict__ K,
    const uint16_t* __restrict__ Vt, uint16_t* __restrict__ Obf)
{
  __shared__ uint16_t ldsK[2][32 * 64];   // [buf][key(0..31)][d(0..63)]  4 KB
  __shared__ uint16_t ldsV[2][64 * 32];   // [buf][d(0..63)][key(0..31)]  4 KB
  __shared__ uint16_t ldsP[4][16 * 32];   // per-wave P tile               1 KB

  const int tid  = threadIdx.x;
  const int wave = tid >> 5;
  const int lane = tid & 31;
  const int ln   = lane & 15;
  const int hf   = lane >> 4;

  const int blocksPerBH = S_N / 64;             // 32
  const int bh     = blockIdx.x / blocksPerBH;  // 0..31
  const int qBlock = (blockIdx.x % blocksPerBH) * 64;
  const int qBase  = qBlock + wave * 16;        // this wave's q-tile
  const int kEnd   = qBase + 16;                // exclusive causal key bound
  const int nT     = qBlock / 32 + 2;           // 32-key tiles for the block

  const uint16_t* Qh  = Q  + (size_t)bh * S_N * HD_N;
  const uint16_t* Kh  = K  + (size_t)bh * S_N * HD_N;
  const uint16_t* Vth = Vt + (size_t)bh * HD_N * S_N;
  uint16_t* P = ldsP[wave];

  // Stage one 32-key K/V tile pair: 256+256 16B chunks over 128 threads.
  auto stage = [&](int buf, int kb) {
    const uint32_t lk = (uint32_t)(uintptr_t)&ldsK[buf][0];
    const uint32_t lv = (uint32_t)(uintptr_t)&ldsV[buf][0];
#pragma unroll
    for (int i = 0; i < 2; ++i) {
      const int ck = tid + i * 128;             // 0..255
      const int krow = ck >> 3, koff = ck & 7;  // K: 32 rows x 8 chunks
      async_cp16(lk + (uint32_t)(krow * 128 + koff * 16),
                 (uint32_t)((kb + krow) * (HD_N * 2) + koff * 16), Kh);
      const int vrow = ck >> 2, voff = ck & 3;  // V: 64 rows x 4 chunks
      async_cp16(lv + (uint32_t)(vrow * 64 + voff * 16),
                 (uint32_t)(vrow * (S_N * 2) + kb * 2 + voff * 16), Vth);
    }
  };

  // Persistent Q A-fragments for k = 0..31 and 32..63
  const uint16_t* qrow = Qh + (size_t)(qBase + ln) * HD_N;
  v16bf qa[2];
#pragma unroll
  for (int kc = 0; kc < 2; ++kc)
    qa[kc] = frag_cat(qrow + kc * 32 + hf * 16, qrow + kc * 32 + hf * 16 + 8);

  v8f o[4] = {};
  float rowMax[8], rowSum[8];
#pragma unroll
  for (int r = 0; r < 8; ++r) { rowMax[r] = NEG_BIG; rowSum[r] = 0.0f; }

  stage(0, 0);
  wait_async0();
  __syncthreads();

  for (int kt = 0; kt < nT; ++kt) {
    const int buf = kt & 1;
    const int kb  = kt * 32;
    if (kt + 1 < nT) stage(buf ^ 1, kb + 32);   // overlap DMA w/ math

    if (kb < kEnd) {   // wave-uniform: this wave still has causal work
      const uint16_t* Ks = ldsK[buf];
      const uint16_t* Vs = ldsV[buf];

      // ---- S = Q K^T : two 16-wide key halves, two 32-deep k chunks ----
      v8f s0 = {}, s1 = {};
#pragma unroll
      for (int kc = 0; kc < 2; ++kc) {
        const uint16_t* k0 = Ks + ln * HD_N + kc * 32 + hf * 16;
        v16bf b0 = frag_cat(k0, k0 + 8);
        s0 = __builtin_amdgcn_wmma_f32_16x16x32_bf16(false, qa[kc], false, b0,
                                                     (short)0, s0, false, false);
        const uint16_t* k1 = Ks + (16 + ln) * HD_N + kc * 32 + hf * 16;
        v16bf b1 = frag_cat(k1, k1 + 8);
        s1 = __builtin_amdgcn_wmma_f32_16x16x32_bf16(false, qa[kc], false, b1,
                                                     (short)0, s1, false, false);
      }

      // ---- causal mask (only diagonal-straddling tiles) ----
      if (kb + 31 > qBase) {
        const int key0 = kb + ln, key1 = kb + 16 + ln;
#pragma unroll
        for (int r = 0; r < 8; ++r) {
          const int row = qBase + r + hf * 8;
          if (key0 > row) s0[r] = NEG_BIG;
          if (key1 > row) s1[r] = NEG_BIG;
        }
      }

      // ---- online softmax: row max across the 16 lanes holding each row ----
      float tmax[8];
#pragma unroll
      for (int r = 0; r < 8; ++r) {
        float t = fmaxf(s0[r], s1[r]);
#pragma unroll
        for (int m = 1; m < 16; m <<= 1) t = fmaxf(t, __shfl_xor(t, m, 32));
        tmax[r] = t;
      }

      float psum[8];
#pragma unroll
      for (int r = 0; r < 8; ++r) {
        const float mNew = fmaxf(rowMax[r], tmax[r]);
        const float corr = __expf(rowMax[r] - mNew);
        rowMax[r] = mNew;
        rowSum[r] *= corr;
#pragma unroll
        for (int j = 0; j < 4; ++j) o[j][r] *= corr;
        const float p0 = __expf(s0[r] - mNew);
        const float p1 = __expf(s1[r] - mNew);
        psum[r] = p0 + p1;
        const int m = r + hf * 8;
        P[m * 32 + ln]      = f32_bf16(p0);   // C-layout -> LDS [row][key]
        P[m * 32 + 16 + ln] = f32_bf16(p1);
      }
#pragma unroll
      for (int r = 0; r < 8; ++r) {
        float t = psum[r];
#pragma unroll
        for (int m = 1; m < 16; m <<= 1) t += __shfl_xor(t, m, 32);
        rowSum[r] += t;
      }

      // ---- reload P in A-fragment layout (per-wave LDS; DS in-order) ----
      const uint16_t* pr = P + ln * 32;
      v16bf pa = frag_cat(pr + hf * 16, pr + hf * 16 + 8);

      // ---- O += P V : V^T tile rows contiguous over keys ----
#pragma unroll
      for (int j = 0; j < 4; ++j) {
        const uint16_t* vrow = Vs + (j * 16 + ln) * 32 + hf * 16;
        v16bf vb = frag_cat(vrow, vrow + 8);
        o[j] = __builtin_amdgcn_wmma_f32_16x16x32_bf16(false, pa, false, vb,
                                                       (short)0, o[j], false, false);
      }
    }

    wait_async0();
    __syncthreads();
  }

  // ---- normalize and store merged-head layout [b][s][h*64+d] (bf16) ----
  const int b = bh / H_N, h = bh % H_N;
#pragma unroll
  for (int j = 0; j < 4; ++j) {
#pragma unroll
    for (int r = 0; r < 8; ++r) {
      const int s = qBase + r + hf * 8;
      const int d = j * 16 + ln;
      const float val = o[j][r] / rowSum[r];
      Obf[((size_t)(b * S_N + s)) * D_N + h * HD_N + d] = f32_bf16(val);
    }
  }
}

// ---------------------------------------------------------------------------
// Host-side orchestration (graph-capture safe: kernel launches only)
// ---------------------------------------------------------------------------
extern "C" void kernel_launch(void* const* d_in, const int* in_sizes, int n_in,
                              void* d_out, int out_size, void* d_ws, size_t ws_size,
                              hipStream_t stream) {
  (void)in_sizes; (void)n_in; (void)out_size; (void)ws_size;
  const float* hs = (const float*)d_in[0];
  const float* Wq = (const float*)d_in[1];
  const float* Wk = (const float*)d_in[2];
  const float* Wv = (const float*)d_in[3];
  const float* Wo = (const float*)d_in[4];
  float* out = (float*)d_out;

  // Workspace layout (48 MiB total)
  char* ws = (char*)d_ws;
  const size_t xbytes = (size_t)M_TOT * D_N * 2;   // 8 MiB
  const size_t wbytes = (size_t)D_N * D_N * 2;     // 2 MiB
  uint16_t* Xbf  = (uint16_t*)(ws);
  uint16_t* Wqb  = (uint16_t*)(ws + xbytes);
  uint16_t* Wkb  = (uint16_t*)(ws + xbytes + wbytes);
  uint16_t* Wvb  = (uint16_t*)(ws + xbytes + 2 * wbytes);
  uint16_t* Wob  = (uint16_t*)(ws + xbytes + 3 * wbytes);
  uint16_t* Qb   = (uint16_t*)(ws + xbytes + 4 * wbytes);
  uint16_t* Kb   = (uint16_t*)(ws + 2 * xbytes + 4 * wbytes);
  uint16_t* Vtb  = (uint16_t*)(ws + 3 * xbytes + 4 * wbytes);
  uint16_t* AOb  = (uint16_t*)(ws + 4 * xbytes + 4 * wbytes);

  const int nX = M_TOT * D_N;     // 4194304
  const int nW = D_N * D_N;       // 1048576
  cvt_bf16_k<<<(nX + 255) / 256, 256, 0, stream>>>(hs, Xbf, nX, 1.0f);
  cvt_bf16_k<<<(nW + 255) / 256, 256, 0, stream>>>(Wq, Wqb, nW, ATT_SCALE); // fold scale
  cvt_bf16_k<<<(nW + 255) / 256, 256, 0, stream>>>(Wk, Wkb, nW, 1.0f);
  cvt_bf16_k<<<(nW + 255) / 256, 256, 0, stream>>>(Wv, Wvb, nW, 1.0f);
  cvt_bf16_k<<<(nW + 255) / 256, 256, 0, stream>>>(Wo, Wob, nW, 1.0f);

  dim3 ggrid(M_TOT / 128, D_N / 128);   // 32 x 8, 128x128 tiles
  gemm_xwT_k<<<ggrid, 256, 0, stream>>>(Xbf, Wqb, Qb,  nullptr, 0); // Q (pre-scaled)
  gemm_xwT_k<<<ggrid, 256, 0, stream>>>(Xbf, Wkb, Kb,  nullptr, 0); // K [b][h][s][d]
  gemm_xwT_k<<<ggrid, 256, 0, stream>>>(Xbf, Wvb, Vtb, nullptr, 1); // V^T [b][h][d][s]

  const int nBlk = B_N * H_N * (S_N / 64);         // 1024 blocks, 4 q-tiles each
  flash_attn_k<<<nBlk, 128, 0, stream>>>(Qb, Kb, Vtb, AOb);

  gemm_xwT_k<<<ggrid, 256, 0, stream>>>(AOb, Wob, nullptr, out, 2); // out = A @ Wo^T (f32)
}
